// DagEncoder_12979391169440
// MI455X (gfx1250) — matrix operand order repaired and checked by reading, not compiled, not run on hardware
//
#include <hip/hip_runtime.h>

typedef __attribute__((ext_vector_type(16))) _Float16 v16h;
typedef __attribute__((ext_vector_type(8)))  _Float16 v8h;
typedef __attribute__((ext_vector_type(8)))  float    v8f;

#define THREADS 256
#define NODES_PER_BLOCK 128

union V16U { v8h h8[2]; v16h v; };

// Load a 16-half fragment as two 16-byte LDS reads.
__device__ inline v16h lds_frag16(const _Float16* p0, const _Float16* p1) {
    V16U u;
    u.h8[0] = *reinterpret_cast<const v8h*>(p0);
    u.h8[1] = *reinterpret_cast<const v8h*>(p1);
    return u.v;
}

__device__ inline v8f wmma_f16(v16h a, v16h b, v8f c) {
    // (neg_a, A, neg_b, B, c_mod, C, reuse_a, reuse_b)
    return __builtin_amdgcn_wmma_f32_16x16x32_f16(false, a, false, b, (short)0, c,
                                                  false, false);
}

extern "C" __global__ void zero_out_kernel(float* __restrict__ out, int n) {
    int i = blockIdx.x * blockDim.x + threadIdx.x;
    if (i < n) out[i] = 0.0f;
}

extern "C" __global__ __launch_bounds__(THREADS)
void dag_mlp_seg_kernel(const float* __restrict__ x,      // [N,8]
                        const float* __restrict__ hn,     // [N,32]
                        const float* __restrict__ W1,     // [40,64]
                        const float* __restrict__ b1,     // [64]
                        const float* __restrict__ W2,     // [64,64]
                        const float* __restrict__ b2,     // [64]
                        const float* __restrict__ W3,     // [64,32]
                        const float* __restrict__ b3,     // [32]
                        const int*   __restrict__ ptr,    // [M+1]
                        float*       __restrict__ out,    // [M,32]
                        int Ntot, int Mseg)
{
    // Weights transposed to [n][k] f16 so B fragments are contiguous in K.
    __shared__ __align__(32) _Float16 sW1[64 * 64];   // k>=40 zero-padded
    __shared__ __align__(32) _Float16 sW2[64 * 64];
    __shared__ __align__(32) _Float16 sW3[32 * 64];
    __shared__ float sB1[64], sB2[64], sB3[32];
    __shared__ __align__(32) _Float16 sSt[8][16 * 64]; // per-wave activation staging
    __shared__ __align__(32) float    sOut[8][16 * 32];// per-wave layer-3 output
    __shared__ float sBucket[128 * 32];                // per-segment partial sums
    __shared__ int   sSegId[NODES_PER_BLOCK];
    __shared__ int   sRank[NODES_PER_BLOCK];
    __shared__ int   sBucketSeg[NODES_PER_BLOCK];
    __shared__ int   sNumBuckets;

    const int t     = threadIdx.x;
    const int w     = t >> 5;        // wave id, 0..7
    const int lane  = t & 31;
    const int m     = lane & 15;     // row within 16x16 tile
    const int hiSel = lane >> 4;     // lane group (K-half select)
    const int nb    = blockIdx.x * NODES_PER_BLOCK;

    // ---- Phase 0a: stage weights/biases into LDS (f32 -> f16, transpose) ----
    for (int idx = t; idx < 64 * 64; idx += THREADS) {
        int n = idx >> 6, k = idx & 63;
        sW1[idx] = (_Float16)((k < 40) ? W1[k * 64 + n] : 0.0f);
        sW2[idx] = (_Float16)W2[k * 64 + n];
    }
    for (int idx = t; idx < 32 * 64; idx += THREADS) {
        int n = idx >> 6, k = idx & 63;
        sW3[idx] = (_Float16)W3[k * 32 + n];
    }
    if (t < 64) { sB1[t] = b1[t]; sB2[t] = b2[t]; }
    if (t < 32) { sB3[t] = b3[t]; }

    // ---- Phase 0b: segment id per node (binary search: last s with ptr[s]<=v) --
    if (t < NODES_PER_BLOCK) {
        int node = nb + t;
        int v = node < Ntot ? node : (Ntot - 1);
        int a = 0, b = Mseg;                  // ptr indices [0, Mseg]
        while (b - a > 1) {
            int mid = (a + b) >> 1;
            if (ptr[mid] <= v) a = mid; else b = mid;
        }
        sSegId[t] = a;
    }
    for (int idx = t; idx < 128 * 32; idx += THREADS) sBucket[idx] = 0.0f;
    __syncthreads();

    // ---- Phase 0c: run-length rank of sorted segment ids -> bucket index ----
    if (t < NODES_PER_BLOCK) {
        int r = 0;
        for (int j = 1; j <= t; ++j) r += (sSegId[j] != sSegId[j - 1]) ? 1 : 0;
        sRank[t] = r;
        if (t == 0 || sSegId[t] != sSegId[t - 1]) sBucketSeg[r] = sSegId[t];
        if (t == NODES_PER_BLOCK - 1) sNumBuckets = r + 1;
    }
    __syncthreads();

    // ---- Per-wave 16-node tile ----
    int node = nb + w * 16 + m;
    if (node >= Ntot) node = Ntot - 1;        // clamp loads; excluded at reduce
    const float* xr = x  + (size_t)node * 8;
    const float* hr = hn + (size_t)node * 32;
    _Float16* st = sSt[w];

    // ================= Layer 1: [16x64pad] x [64pad x 64] =================
    v8f acc1[4];
    #pragma unroll
    for (int nt = 0; nt < 4; ++nt)
        #pragma unroll
        for (int i = 0; i < 8; ++i) acc1[nt][i] = 0.0f;

    {
        // k-step 0: K 0..31. A runs per lane group map to contiguous global rows.
        const float* r0 = hiSel ? (hr + 8)  : xr;        // K 16..23 | 0..7
        const float* r1 = hiSel ? (hr + 16) : hr;        // K 24..31 | 8..15
        v16h a0;
        #pragma unroll
        for (int i = 0; i < 8; ++i) {
            a0[i]     = (_Float16)r0[i];
            a0[8 + i] = (_Float16)r1[i];
        }
        #pragma unroll
        for (int nt = 0; nt < 4; ++nt) {
            const _Float16* bp = &sW1[(nt * 16 + m) * 64 + hiSel * 16];
            acc1[nt] = wmma_f16(a0, lds_frag16(bp, bp + 8), acc1[nt]);
        }

        // k-step 1: K 32..63 (only K 32..39 nonzero, in lanes 0..15 run 0)
        v16h a1;
        #pragma unroll
        for (int i = 0; i < 16; ++i) a1[i] = (_Float16)0.0f;
        if (!hiSel) {
            #pragma unroll
            for (int i = 0; i < 8; ++i) a1[i] = (_Float16)hr[24 + i];
        }
        #pragma unroll
        for (int nt = 0; nt < 4; ++nt) {
            const _Float16* bp = &sW1[(nt * 16 + m) * 64 + 32 + hiSel * 16];
            acc1[nt] = wmma_f16(a1, lds_frag16(bp, bp + 8), acc1[nt]);
        }
    }
    // bias + ReLU -> f16 staging (C layout: VGPR r -> row r + 8*hiSel, col = nt*16+m)
    #pragma unroll
    for (int nt = 0; nt < 4; ++nt) {
        int col = nt * 16 + m;
        float bias = sB1[col];
        #pragma unroll
        for (int r = 0; r < 8; ++r) {
            float v = acc1[nt][r] + bias;
            st[(r + hiSel * 8) * 64 + col] = (_Float16)(v > 0.0f ? v : 0.0f);
        }
    }
    __syncthreads();

    // ================= Layer 2: [16x64] x [64x64] =================
    v8f acc2[4];
    #pragma unroll
    for (int nt = 0; nt < 4; ++nt)
        #pragma unroll
        for (int i = 0; i < 8; ++i) acc2[nt][i] = 0.0f;

    #pragma unroll
    for (int ks = 0; ks < 2; ++ks) {
        const _Float16* ap = &st[m * 64 + ks * 32 + hiSel * 16];
        v16h a = lds_frag16(ap, ap + 8);
        #pragma unroll
        for (int nt = 0; nt < 4; ++nt) {
            const _Float16* bp = &sW2[(nt * 16 + m) * 64 + ks * 32 + hiSel * 16];
            acc2[nt] = wmma_f16(a, lds_frag16(bp, bp + 8), acc2[nt]);
        }
    }
    __syncthreads();
    #pragma unroll
    for (int nt = 0; nt < 4; ++nt) {
        int col = nt * 16 + m;
        float bias = sB2[col];
        #pragma unroll
        for (int r = 0; r < 8; ++r) {
            float v = acc2[nt][r] + bias;
            st[(r + hiSel * 8) * 64 + col] = (_Float16)(v > 0.0f ? v : 0.0f);
        }
    }
    __syncthreads();

    // ================= Layer 3: [16x64] x [64x32] =================
    v8f acc3[2];
    #pragma unroll
    for (int nt = 0; nt < 2; ++nt)
        #pragma unroll
        for (int i = 0; i < 8; ++i) acc3[nt][i] = 0.0f;

    #pragma unroll
    for (int ks = 0; ks < 2; ++ks) {
        const _Float16* ap = &st[m * 64 + ks * 32 + hiSel * 16];
        v16h a = lds_frag16(ap, ap + 8);
        #pragma unroll
        for (int nt = 0; nt < 2; ++nt) {
            const _Float16* bp = &sW3[(nt * 16 + m) * 64 + ks * 32 + hiSel * 16];
            acc3[nt] = wmma_f16(a, lds_frag16(bp, bp + 8), acc3[nt]);
        }
    }
    float* o = sOut[w];
    #pragma unroll
    for (int nt = 0; nt < 2; ++nt) {
        int col = nt * 16 + m;
        float bias = sB3[col];
        #pragma unroll
        for (int r = 0; r < 8; ++r)
            o[(r + hiSel * 8) * 32 + col] = acc3[nt][r] + bias;
    }
    __syncthreads();

    // ---- Segment reduction: rows -> LDS buckets (ds f32 atomics) ----
    {
        const int c = lane;                       // column 0..31
        #pragma unroll 4
        for (int mm = 0; mm < 16; ++mm) {
            int li = w * 16 + mm;
            if (nb + li < Ntot)
                atomicAdd(&sBucket[sRank[li] * 32 + c], o[mm * 32 + c]);
        }
    }
    __syncthreads();

    // ---- Flush live buckets with global f32 atomics ----
    int nbk = sNumBuckets;
    for (int idx = t; idx < nbk * 32; idx += THREADS) {
        int r = idx >> 5, c = idx & 31;
        atomicAdd(&out[(size_t)sBucketSeg[r] * 32 + c], sBucket[idx]);
    }
}

extern "C" void kernel_launch(void* const* d_in, const int* in_sizes, int n_in,
                              void* d_out, int out_size, void* d_ws, size_t ws_size,
                              hipStream_t stream) {
    const float* x  = (const float*)d_in[0];
    const float* hn = (const float*)d_in[1];
    const float* W1 = (const float*)d_in[2];
    const float* b1 = (const float*)d_in[3];
    const float* W2 = (const float*)d_in[4];
    const float* b2 = (const float*)d_in[5];
    const float* W3 = (const float*)d_in[6];
    const float* b3 = (const float*)d_in[7];
    const int*   ptr = (const int*)d_in[8];
    float* out = (float*)d_out;

    const int Ntot = in_sizes[0] / 8;      // x is [N, 8]
    const int Mseg = in_sizes[8] - 1;      // ptr is [M+1]

    // out must start at zero every launch (atomic accumulation).
    zero_out_kernel<<<(out_size + THREADS - 1) / THREADS, THREADS, 0, stream>>>(out, out_size);

    int nBlocks = (Ntot + NODES_PER_BLOCK - 1) / NODES_PER_BLOCK;
    dag_mlp_seg_kernel<<<nBlocks, THREADS, 0, stream>>>(
        x, hn, W1, b1, W2, b2, W3, b3, ptr, out, Ntot, Mseg);
}
